// RegionalCodebook_40535901339809
// MI455X (gfx1250) — matrix operand order repaired and checked by reading, not compiled, not run on hardware
//
#include <hip/hip_runtime.h>
#include <stdint.h>

// ---------------- types ----------------
typedef __attribute__((ext_vector_type(16))) __bf16 v16bf;
typedef __attribute__((ext_vector_type(8)))  float  v8f;
typedef __attribute__((ext_vector_type(4)))  unsigned int u32x4;
typedef __attribute__((ext_vector_type(4)))  float  f32x4;

union FragBF { u32x4 q[2]; v16bf v; };

#define B_   8
#define N_   4096
#define D_   1024
#define R_   8
#define KR_  16
#define DR_  256
#define SCALE_ 0.03125f   // 1/sqrt(1024)

// ---------------- LDS helpers ----------------
typedef __attribute__((address_space(3))) unsigned char lds_byte;

static __device__ __forceinline__ uint32_t lds_off32(const void* p) {
    // generic -> LDS addrspace cast; ptrtoint of AS(3) pointer = wave-relative LDS byte offset
    return (uint32_t)(uintptr_t)(const lds_byte*)p;
}

// async global -> LDS copy of 16 bytes (per lane), tracked by ASYNCcnt
static __device__ __forceinline__ void async_g2l_b128(uint32_t ldsoff, const void* g) {
    asm volatile("global_load_async_to_lds_b128 %0, %1, off"
                 :: "v"(ldsoff), "v"((uint64_t)(uintptr_t)g)
                 : "memory");
}

static __device__ __forceinline__ void wait_async0() {
    asm volatile("s_wait_asynccnt 0x0" ::: "memory");
}

// ---------------- fp32 -> bf16 (round to nearest even) ----------------
__device__ __forceinline__ uint16_t f2bf(float f) {
    uint32_t u = __builtin_bit_cast(uint32_t, f);
    uint32_t r = u + 0x7FFFu + ((u >> 16) & 1u);
    return (uint16_t)(r >> 16);
}

__global__ void cvt_bf16_kernel(const float* __restrict__ in, uint16_t* __restrict__ outp, int n) {
    int i = blockIdx.x * blockDim.x + threadIdx.x;
    int stride = gridDim.x * blockDim.x;
    for (; i < n; i += stride) outp[i] = f2bf(in[i]);
}

// ---------------- big GEMM: OUT[m,e] = sum_d X[m,d]*W[e,d] (bf16 in, f32 out) ----
// block = 256 threads (8 waves), tile = 128(M) x 64(N), K-step = 32.
// Double-buffered LDS staged via GLOBAL_LOAD_ASYNC_TO_LDS_B128 (ASYNCcnt).
// blockIdx.z: 0 -> W_k/keys, 1 -> W_v/values
__global__ void __launch_bounds__(256)
gemm_kv_kernel(const uint16_t* __restrict__ Xbf,
               const uint16_t* __restrict__ Wkbf,
               const uint16_t* __restrict__ Wvbf,
               float* __restrict__ keys, float* __restrict__ values) {
    const uint16_t* W   = (blockIdx.z == 0) ? Wkbf : Wvbf;
    float*          OUT = (blockIdx.z == 0) ? keys : values;
    const int e0   = blockIdx.x * 64;
    const int m0   = blockIdx.y * 128;
    const int tid  = threadIdx.x;
    const int lane = tid & 31;
    const int wave = tid >> 5;
    const int hi   = lane >> 4;       // lane-half selector
    const int l15  = lane & 15;

    // padded rows: 4 data u32x4 (64B = 32 bf16) + 1 pad u32x4 per row
    __shared__ u32x4 Ash[2][128 * 5];
    __shared__ u32x4 Bsh[2][64 * 5];

    const uint32_t ashOff = lds_off32(&Ash[0][0]);
    const uint32_t bshOff = lds_off32(&Bsh[0][0]);

    // per-thread staging coordinates
    const int arow0 = tid >> 2,  aseg = tid & 3;            // A chunk 0 (rows 0..63)
    const int arow1 = arow0 + 64;                           // A chunk 1 (rows 64..127)
    const int brow  = tid >> 2,  bseg = tid & 3;            // B (rows 0..63)

    const uint16_t* aG0 = Xbf + (size_t)(m0 + arow0) * D_ + aseg * 8;
    const uint16_t* aG1 = Xbf + (size_t)(m0 + arow1) * D_ + aseg * 8;
    const uint16_t* bG  = W   + (size_t)(e0 + brow)  * D_ + bseg * 8;
    const uint32_t aL0 = ashOff + (uint32_t)(arow0 * 5 + aseg) * 16u;
    const uint32_t aL1 = ashOff + (uint32_t)(arow1 * 5 + aseg) * 16u;
    const uint32_t bL  = bshOff + (uint32_t)(brow  * 5 + bseg) * 16u;
    const uint32_t ABUF = 128 * 5 * 16;   // bytes per A buffer
    const uint32_t BBUF = 64 * 5 * 16;    // bytes per B buffer

    v8f c[4];
    const v8f vzero = {0.f,0.f,0.f,0.f,0.f,0.f,0.f,0.f};
    #pragma unroll
    for (int nt = 0; nt < 4; ++nt) c[nt] = vzero;

    // prologue: stage K-step 0 into buffer 0
    async_g2l_b128(aL0, aG0);
    async_g2l_b128(aL1, aG1);
    async_g2l_b128(bL,  bG);

    const int NSTEP = D_ / 32;
    for (int s = 0; s < NSTEP; ++s) {
        const int cur = s & 1;
        wait_async0();          // my async loads into buf[cur] are done
        __syncthreads();        // everyone's are done; prev buf reads also done

        // prefetch next K-step into the other buffer
        if (s + 1 < NSTEP) {
            const int nxt = 1 - cur;
            const int k = (s + 1) * 32;
            async_g2l_b128(aL0 + nxt * ABUF, aG0 + k);
            async_g2l_b128(aL1 + nxt * ABUF, aG1 + k);
            async_g2l_b128(bL  + nxt * BBUF, bG  + k);
        }

        // fragment loads (all issued before the WMMA burst)
        const u32x4* Ac = &Ash[cur][0];
        const u32x4* Bc = &Bsh[cur][0];
        FragBF a, b[4];
        {   // A 16x32: K chunks {8*hi .. 8*hi+7} and {+16}
            const u32x4* ap = &Ac[(wave * 16 + l15) * 5 + hi];
            a.q[0] = ap[0];
            a.q[1] = ap[2];
        }
        #pragma unroll
        for (int nt = 0; nt < 4; ++nt) {   // B 32x16: K chunks {8*hi .. 8*hi+7} and {+16}
            const u32x4* bp = &Bc[(nt * 16 + l15) * 5 + hi];
            b[nt].q[0] = bp[0];
            b[nt].q[1] = bp[2];
        }
        #pragma unroll
        for (int nt = 0; nt < 4; ++nt)
            c[nt] = __builtin_amdgcn_wmma_f32_16x16x32_bf16(
                        false, a.v, false, b[nt].v, (short)0, c[nt], false, false);
        // no trailing barrier: each wave's ds_loads complete (dscnt-waited) before
        // its WMMAs, so the barrier at the top of the next step is sufficient.
    }

    // store C: VGPR g -> M = g + 8*hi, N = l15
    #pragma unroll
    for (int nt = 0; nt < 4; ++nt) {
        #pragma unroll
        for (int g = 0; g < 8; ++g) {
            int m = m0 + wave * 16 + g + hi * 8;
            int n = e0 + nt * 16 + l15;
            OUT[(size_t)m * D_ + n] = c[nt][g];
        }
    }
}

// ---------------- q = region_tokens @ W_q^T  (R x D) ----------------
__global__ void qproj_kernel(const float* __restrict__ region_tokens,
                             const float* __restrict__ W_q,
                             float* __restrict__ qbuf) {
    int idx = blockIdx.x * 256 + threadIdx.x;      // 8192 outputs
    int r = idx >> 10, e = idx & 1023;
    const f32x4* a = (const f32x4*)(region_tokens + (size_t)r * D_);
    const f32x4* w = (const f32x4*)(W_q + (size_t)e * D_);
    float acc = 0.f;
    for (int d = 0; d < D_ / 4; ++d) {
        f32x4 x = a[d], y = w[d];
        acc += x.x * y.x + x.y * y.y + x.z * y.z + x.w * y.w;
    }
    qbuf[idx] = acc;
}

// ---------------- scores[b,r,n] = scale * q[r] . keys[b,n] (masked) ------------
__global__ void scores_kernel(const float* __restrict__ qbuf,
                              const float* __restrict__ keys,
                              const unsigned char* __restrict__ mask,
                              float* __restrict__ attn) {
    int b = blockIdx.z, r = blockIdx.y;
    int n = blockIdx.x * 256 + threadIdx.x;
    const f32x4* qv = (const f32x4*)(qbuf + (size_t)r * D_);
    const f32x4* kv = (const f32x4*)(keys + ((size_t)b * N_ + n) * D_);
    float acc = 0.f;
    for (int d = 0; d < D_ / 4; ++d) {
        f32x4 x = qv[d], y = kv[d];
        acc += x.x * y.x + x.y * y.y + x.z * y.z + x.w * y.w;
    }
    float s = acc * SCALE_;
    if (!mask[(size_t)b * N_ + n]) s = -INFINITY;
    attn[((size_t)(b * R_ + r)) * N_ + n] = s;
}

// ---------------- softmax over N per (b,r), nan_to_num semantics --------------
__global__ void softmax_n_kernel(float* __restrict__ attn) {
    int r = blockIdx.x, b = blockIdx.y, tid = threadIdx.x;
    float* s = attn + ((size_t)(b * R_ + r)) * N_;
    __shared__ float red[256];
    float m = -INFINITY;
    for (int n = tid; n < N_; n += 256) m = fmaxf(m, s[n]);
    red[tid] = m; __syncthreads();
    for (int o = 128; o > 0; o >>= 1) {
        if (tid < o) red[tid] = fmaxf(red[tid], red[tid + o]);
        __syncthreads();
    }
    m = red[0]; __syncthreads();
    bool ok = (m > -INFINITY);
    float sum = 0.f;
    for (int n = tid; n < N_; n += 256) sum += ok ? __expf(s[n] - m) : 0.f;
    red[tid] = sum; __syncthreads();
    for (int o = 128; o > 0; o >>= 1) {
        if (tid < o) red[tid] += red[tid + o];
        __syncthreads();
    }
    sum = red[0];
    float inv = (ok && sum > 0.f) ? 1.f / sum : 0.f;
    for (int n = tid; n < N_; n += 256) s[n] = ok ? __expf(s[n] - m) * inv : 0.f;
}

// ---------------- rf_pre[b,r,e] = sum_n attn[b,r,n] * values[b,n,e] -----------
__global__ void rfpre_kernel(const float* __restrict__ attn,
                             const float* __restrict__ values,
                             float* __restrict__ rfp) {
    int b = blockIdx.z, r = blockIdx.y;
    int e = blockIdx.x * 256 + threadIdx.x;
    const float* a = attn + ((size_t)(b * R_ + r)) * N_;
    const float* v = values + (size_t)b * N_ * D_ + e;
    float acc = 0.f;
    for (int n = 0; n < N_; ++n) acc += a[n] * v[(size_t)n * D_];
    rfp[((size_t)(b * R_ + r)) * D_ + e] = acc;
}

// ---------------- rf[b,r,e] = rf_pre[b,r] . W_out[e] --------------------------
__global__ void rfout_kernel(const float* __restrict__ rfp,
                             const float* __restrict__ W_out,
                             float* __restrict__ rf) {
    int b = blockIdx.z, r = blockIdx.y;
    int e = blockIdx.x * 256 + threadIdx.x;
    const f32x4* a = (const f32x4*)(rfp + ((size_t)(b * R_ + r)) * D_);
    const f32x4* w = (const f32x4*)(W_out + (size_t)e * D_);
    float acc = 0.f;
    for (int d = 0; d < D_ / 4; ++d) {
        f32x4 x = a[d], y = w[d];
        acc += x.x * y.x + x.y * y.y + x.z * y.z + x.w * y.w;
    }
    rf[((size_t)(b * R_ + r)) * D_ + e] = acc;
}

// ---------------- routing: logits, softmax(K_R), modes = (w@codebook)@W_oproj^T
__global__ void route_kernel(const float* __restrict__ rf,
                             const float* __restrict__ W_rout,
                             const float* __restrict__ b_rout,
                             const float* __restrict__ codebook,
                             const float* __restrict__ W_oproj,
                             float* __restrict__ modes,
                             float* __restrict__ out_logits) {
    int r = blockIdx.x, b = blockIdx.y, tid = threadIdx.x;
    const float* x = rf + ((size_t)(b * R_ + r)) * D_;
    __shared__ float lg[KR_], w[KR_], tmp[DR_];
    if (tid < KR_) {
        float acc = b_rout[tid];
        const float* wr = W_rout + (size_t)tid * D_;
        for (int d = 0; d < D_; ++d) acc += x[d] * wr[d];
        lg[tid] = acc;
        out_logits[((size_t)(b * R_ + r)) * KR_ + tid] = acc;
    }
    __syncthreads();
    if (tid == 0) {
        float m = -INFINITY;
        for (int k = 0; k < KR_; ++k) m = fmaxf(m, lg[k]);
        float s = 0.f;
        for (int k = 0; k < KR_; ++k) { float e = __expf(lg[k] - m); w[k] = e; s += e; }
        float inv = 1.f / s;
        for (int k = 0; k < KR_; ++k) w[k] *= inv;
    }
    __syncthreads();
    { // tmp[j] = sum_k w[k]*codebook[k,j]
        float acc = 0.f;
        for (int k = 0; k < KR_; ++k) acc += w[k] * codebook[(size_t)k * DR_ + tid];
        tmp[tid] = acc;
    }
    __syncthreads();
    { // modes[e] = sum_j tmp[j]*W_oproj[e,j]
        float acc = 0.f;
        const float* wo = W_oproj + (size_t)tid * DR_;
        for (int j = 0; j < DR_; ++j) acc += tmp[j] * wo[j];
        modes[((size_t)(b * R_ + r)) * DR_ + tid] = acc;
    }
}

// ---------------- node->region softmax + mode mixing --------------------------
__global__ void node_out_kernel(const float* __restrict__ keys,
                                const float* __restrict__ rf,
                                const float* __restrict__ modes,
                                float* __restrict__ out) {
    int n = blockIdx.x, b = blockIdx.y, tid = threadIdx.x;
    const float* kvec = keys + ((size_t)b * N_ + n) * D_;
    __shared__ float sh[256 * R_];
    __shared__ float p[R_];
    float acc[R_];
    #pragma unroll
    for (int r = 0; r < R_; ++r) acc[r] = 0.f;
    for (int d = tid; d < D_; d += 256) {
        float kv = kvec[d];
        #pragma unroll
        for (int r = 0; r < R_; ++r)
            acc[r] += kv * rf[((size_t)(b * R_ + r)) * D_ + d];
    }
    #pragma unroll
    for (int r = 0; r < R_; ++r) sh[tid + 256 * r] = acc[r];
    __syncthreads();
    for (int o = 128; o > 0; o >>= 1) {
        if (tid < o) {
            #pragma unroll
            for (int r = 0; r < R_; ++r) sh[tid + 256 * r] += sh[tid + o + 256 * r];
        }
        __syncthreads();
    }
    if (tid == 0) {
        float s[R_], m = -INFINITY;
        #pragma unroll
        for (int r = 0; r < R_; ++r) { s[r] = sh[256 * r] * SCALE_; m = fmaxf(m, s[r]); }
        float sum = 0.f;
        #pragma unroll
        for (int r = 0; r < R_; ++r) { float e = __expf(s[r] - m); p[r] = e; sum += e; }
        float inv = 1.f / sum;
        #pragma unroll
        for (int r = 0; r < R_; ++r) p[r] *= inv;
    }
    __syncthreads();
    float o = 0.f;
    #pragma unroll
    for (int r = 0; r < R_; ++r)
        o += p[r] * modes[((size_t)(b * R_ + r)) * DR_ + tid];
    out[((size_t)b * N_ + n) * DR_ + tid] = o;
}

// ---------------- launch ----------------
extern "C" void kernel_launch(void* const* d_in, const int* in_sizes, int n_in,
                              void* d_out, int out_size, void* d_ws, size_t ws_size,
                              hipStream_t stream) {
    (void)in_sizes; (void)n_in; (void)out_size; (void)ws_size;
    const float*         carry         = (const float*)d_in[0];
    const unsigned char* mask          = (const unsigned char*)d_in[1];
    const float*         codebook      = (const float*)d_in[2];
    const float*         region_tokens = (const float*)d_in[3];
    const float*         W_q           = (const float*)d_in[4];
    const float*         W_k           = (const float*)d_in[5];
    const float*         W_v           = (const float*)d_in[6];
    const float*         W_out         = (const float*)d_in[7];
    const float*         W_rout        = (const float*)d_in[8];
    const float*         b_rout       = (const float*)d_in[9];
    const float*         W_oproj       = (const float*)d_in[10];
    float* out = (float*)d_out;
    char*  ws  = (char*)d_ws;

    // workspace carving (bytes)
    uint16_t* Xbf    = (uint16_t*)(ws + 0);                 //  64 MB  bf16 X
    uint16_t* Wkbf   = (uint16_t*)(ws + 67108864ull);       //   2 MB
    uint16_t* Wvbf   = (uint16_t*)(ws + 69206016ull);       //   2 MB
    float*    keys   = (float*)(ws + 71303168ull);          // 128 MB
    float*    values = (float*)(ws + 205520896ull);         // 128 MB
    float*    qbuf   = (float*)(ws + 339738624ull);         //  32 KB
    float*    attn   = (float*)(ws + 339771392ull);         //   1 MB
    float*    rfp    = (float*)(ws + 340819968ull);         // 256 KB
    float*    rf     = (float*)(ws + 341082112ull);         // 256 KB
    float*    modes  = (float*)(ws + 341344256ull);         //  64 KB

    // 1) bf16 conversions
    cvt_bf16_kernel<<<4096, 256, 0, stream>>>(carry, Xbf, B_ * N_ * D_);
    cvt_bf16_kernel<<<512,  256, 0, stream>>>(W_k, Wkbf, D_ * D_);
    cvt_bf16_kernel<<<512,  256, 0, stream>>>(W_v, Wvbf, D_ * D_);
    // 2) big WMMA GEMM: keys & values
    gemm_kv_kernel<<<dim3(D_ / 64, (B_ * N_) / 128, 2), 256, 0, stream>>>(
        Xbf, Wkbf, Wvbf, keys, values);
    // 3) q projection (R x D)
    qproj_kernel<<<(R_ * D_) / 256, 256, 0, stream>>>(region_tokens, W_q, qbuf);
    // 4) region-token attention over nodes
    scores_kernel<<<dim3(N_ / 256, R_, B_), 256, 0, stream>>>(qbuf, keys, mask, attn);
    softmax_n_kernel<<<dim3(R_, B_), 256, 0, stream>>>(attn);
    rfpre_kernel<<<dim3(D_ / 256, R_, B_), 256, 0, stream>>>(attn, values, rfp);
    rfout_kernel<<<dim3(D_ / 256, R_, B_), 256, 0, stream>>>(rfp, W_out, rf);
    // 5) codebook routing (+ logits into tail of d_out)
    route_kernel<<<dim3(R_, B_), 256, 0, stream>>>(rf, W_rout, b_rout, codebook,
                                                   W_oproj, modes,
                                                   out + (size_t)B_ * N_ * DR_);
    // 6) node->region softmax + mode output
    node_out_kernel<<<dim3(N_, B_), 256, 0, stream>>>(keys, rf, modes, out);
}
